// DualGCN_36636071035174
// MI455X (gfx1250) — matrix operand order repaired and checked by reading, not compiled, not run on hardware
//
#include <hip/hip_runtime.h>
#include <hip/hip_bf16.h>

// ---------- types ----------
typedef __attribute__((ext_vector_type(16))) __bf16       v16bf;
typedef __attribute__((ext_vector_type(8)))  float        v8f;
typedef __attribute__((ext_vector_type(4)))  unsigned int u32x4;

union Frag16 { v16bf v; u32x4 q[2]; };
union Pack8  { u32x4 q; __bf16 e[8]; };

#define BM 128
#define BN 128
#define BK 32
#define LDSA 40   // 32 + 8 pad (80B row stride -> conflict-free)
#define LDSB 40

// ---------------------------------------------------------------------------
// bf16 WMMA GEMM: C[M,N] = A[M,K]*B[K,N] (+bias), A/B bf16 row-major, C f32.
// Requires: N % 128 == 0, K % 32 == 0. M guarded (row-clamped loads, guarded
// stores: out-of-range tile rows compute garbage that is never stored).
// Double-buffered LDS; A tile staged with GLOBAL_LOAD_ASYNC_TO_LDS_B128
// (ASYNCcnt), B tile staged via VGPRs with an LDS transpose.
// ---------------------------------------------------------------------------
__global__ __launch_bounds__(256)
void gemm_bf16_wmma(const __bf16* __restrict__ A, const __bf16* __restrict__ B,
                    float* __restrict__ C, const float* __restrict__ bias,
                    int M, int N, int K)
{
    __shared__ __bf16 As [2][BM][LDSA];    // row-major A tiles
    __shared__ __bf16 BsT[2][BN][LDSB];    // transposed B tiles: BsT[n][k]

    const int tid  = threadIdx.x;
    const int wave = tid >> 5;
    const int lane = tid & 31;
    const int lm   = lane & 15;
    const int half = lane >> 4;

    const int m0 = blockIdx.y * BM;
    const int n0 = blockIdx.x * BN;
    const int wm = (wave & 3) * 32;    // wave grid 4 (M) x 2 (N)
    const int wn = (wave >> 2) * 64;

    v8f acc[2][4];
#pragma unroll
    for (int i = 0; i < 2; i++)
#pragma unroll
        for (int j = 0; j < 4; j++)
#pragma unroll
            for (int r = 0; r < 8; r++) acc[i][j][r] = 0.0f;

    // global->LDS load assignments
    const int la_r = tid >> 1;            // 0..127 (A row in tile)
    const int la_h = (tid & 1) * 16;      // 0 / 16  (A k-half)
    const int lb_k = tid & 31;            // 0..31   (B k row)
    const int lb_n = (tid >> 5) * 16;     // 0..112  (B n group)

    int gmA = m0 + la_r;
    if (gmA > M - 1) gmA = M - 1;                       // clamp: safe, see above
    const __bf16* gA = A + (size_t)gmA * K + la_h;      // + k0 at use
    const __bf16* gB = B + (size_t)lb_k * N + n0 + lb_n; // + k0*N at use

    // ---- tile staging helpers ----
    auto stageA_async = [&](int k0, int buf) {
        unsigned lds0 = (unsigned)(uintptr_t)&As[buf][la_r][la_h];
        const __bf16* p = gA + k0;
        // INST_OFFSET applies to both LDS and global address (ISA 10.7 / 15.18):
        // two 16B transfers cover this thread's 32B slice of the A tile.
        asm volatile("global_load_async_to_lds_b128 %0, %1, off\n\t"
                     "global_load_async_to_lds_b128 %0, %1, off offset:16"
                     :: "v"(lds0), "v"(p) : "memory");
    };
    auto stageB = [&](int k0, int buf) {
        const __bf16* p = gB + (size_t)k0 * N;
        Pack8 t0, t1;
        t0.q = *(const u32x4*)p;
        t1.q = *(const u32x4*)(p + 8);
#pragma unroll
        for (int i = 0; i < 8; i++) BsT[buf][lb_n + i][lb_k]     = t0.e[i];
#pragma unroll
        for (int i = 0; i < 8; i++) BsT[buf][lb_n + 8 + i][lb_k] = t1.e[i];
    };

    // ---- prologue: stage tile 0 into buffer 0 ----
    stageA_async(0, 0);
    stageB(0, 0);
    asm volatile("s_wait_asynccnt 0x0" ::: "memory");
    __syncthreads();

    const int T = K / BK;
    for (int t = 0; t < T; ++t) {
        const int cur = t & 1;
        const int nxt = cur ^ 1;
        const int k0  = t * BK;

        if (t + 1 < T) {
            // prefetch 2 tiles ahead into GL2 (global_prefetch_b8)
            if (t + 2 < T) {
                __builtin_prefetch(gA + (size_t)(t + 2) * BK, 0, 1);
                __builtin_prefetch(gB + (size_t)(t + 2) * BK * N, 0, 1);
            }
            // stage next tile into the other buffer (overlaps with WMMAs below;
            // that buffer was last read at iteration t-1, barrier already passed)
            stageA_async(k0 + BK, nxt);
            stageB(k0 + BK, nxt);
        }

        // ---- fragments per CDNA5 16-bit WMMA layouts ----
        // A 16x32: lane half -> elems 0..7 = K[8h..8h+7], elems 8..15 = K[16+8h..]
        Frag16 a[2];
#pragma unroll
        for (int i = 0; i < 2; i++) {
            const __bf16* base = &As[cur][wm + i * 16 + lm][0];
            a[i].q[0] = *(const u32x4*)(base + 8 * half);
            a[i].q[1] = *(const u32x4*)(base + 16 + 8 * half);
        }
        // B 32x16: lane half -> elems 0..7 = K[8h..8h+7], elems 8..15 = K[16+8h..] of col n
        Frag16 b[4];
#pragma unroll
        for (int j = 0; j < 4; j++) {
            const __bf16* base = &BsT[cur][wn + j * 16 + lm][0];
            b[j].q[0] = *(const u32x4*)(base + 8 * half);
            b[j].q[1] = *(const u32x4*)(base + 16 + 8 * half);
        }
#pragma unroll
        for (int i = 0; i < 2; i++)
#pragma unroll
            for (int j = 0; j < 4; j++)
                acc[i][j] = __builtin_amdgcn_wmma_f32_16x16x32_bf16(
                    false, a[i].v, false, b[j].v, (short)0, acc[i][j], false, false);

        // next-tile async A copies must be resident before the next iteration
        asm volatile("s_wait_asynccnt 0x0" ::: "memory");
        __syncthreads();
    }

    // ---- store C (layout: VGPR r -> M = r + 8*half, N = lane%16) ----
#pragma unroll
    for (int j = 0; j < 4; j++) {
        const int n  = n0 + wn + j * 16 + lm;
        const float bv = bias ? bias[n] : 0.0f;
#pragma unroll
        for (int i = 0; i < 2; i++) {
#pragma unroll
            for (int r = 0; r < 8; r++) {
                const int m = m0 + wm + i * 16 + r + 8 * half;
                if (m < M) C[(size_t)m * N + n] = acc[i][j][r] + bv;
            }
        }
    }
}

// ---------------------------------------------------------------------------
// f32 -> bf16 with row/col zero-padding and optional ReLU.
// dst is [drows, dcols]; src is [srows, scols].
// ---------------------------------------------------------------------------
__global__ void cvt_pad_bf16(const float* __restrict__ src, __bf16* __restrict__ dst,
                             int drows, int srows, int scols, int dcols, int relu)
{
    size_t idx = (size_t)blockIdx.x * blockDim.x + threadIdx.x;
    size_t total = (size_t)drows * dcols;
    if (idx >= total) return;
    int r = (int)(idx / dcols);
    int c = (int)(idx % dcols);
    float v = 0.0f;
    if (r < srows && c < scols) v = src[(size_t)r * scols + c];
    if (relu && v < 0.0f) v = 0.0f;
    dst[idx] = (__bf16)v;
}

// concat [a | b] along features -> bf16
__global__ void cvt_cat_bf16(const float* __restrict__ a, const float* __restrict__ b,
                             __bf16* __restrict__ dst, int Nn, int O)
{
    size_t idx = (size_t)blockIdx.x * blockDim.x + threadIdx.x;
    size_t total = (size_t)Nn * (2 * O);
    if (idx >= total) return;
    int r = (int)(idx / (2 * O));
    int c = (int)(idx % (2 * O));
    float v = (c < O) ? a[(size_t)r * O + c] : b[(size_t)r * O + (c - O)];
    dst[idx] = (__bf16)v;
}

// ---------------------------------------------------------------------------
// Graph kernels (scatter-add GCN aggregation)
// ---------------------------------------------------------------------------
__global__ void fill_ones(float* __restrict__ p, int n)
{
    int i = blockIdx.x * blockDim.x + threadIdx.x;
    if (i < n) p[i] = 1.0f;   // self-loop weight
}

__global__ void deg_acc(const int* __restrict__ col, const float* __restrict__ ew,
                        float* __restrict__ deg, int E)
{
    int e = blockIdx.x * blockDim.x + threadIdx.x;
    if (e < E) atomicAdd(&deg[col[e]], ew[e]);
}

__global__ void make_dinv(const float* __restrict__ deg, float* __restrict__ dinv, int n)
{
    int i = blockIdx.x * blockDim.x + threadIdx.x;
    if (i < n) {
        float d = deg[i];
        dinv[i] = d > 0.0f ? rsqrtf(d) : 0.0f;
    }
}

// out[i,f] = bias[f] + h[i,f] * dinv[i]^2     (self-loop + bias, no atomics)
__global__ void agg_init(const float* __restrict__ h, const float* __restrict__ dinv,
                         const float* __restrict__ bias, float* __restrict__ out,
                         int Nn, int F4)
{
    size_t idx = (size_t)blockIdx.x * blockDim.x + threadIdx.x;
    size_t total = (size_t)Nn * F4;
    if (idx >= total) return;
    int i  = (int)(idx / F4);
    int f4 = (int)(idx % F4);
    float di = dinv[i];
    float s  = di * di;
    float4 hv = ((const float4*)h)[idx];
    float4 o;
    o.x = bias[f4 * 4 + 0] + hv.x * s;
    o.y = bias[f4 * 4 + 1] + hv.y * s;
    o.z = bias[f4 * 4 + 2] + hv.z * s;
    o.w = bias[f4 * 4 + 3] + hv.w * s;
    ((float4*)out)[idx] = o;
}

// one block per edge; block = F/4 threads; out[col] += h[row] * norm
__global__ void agg_scatter(const float* __restrict__ h, const int* __restrict__ row,
                            const int* __restrict__ col, const float* __restrict__ ew,
                            const float* __restrict__ dinv, float* __restrict__ out,
                            int E, int F4)
{
    int e = blockIdx.x;
    if (e >= E) return;
    int r = row[e];
    int c = col[e];
    float norm = dinv[r] * ew[e] * dinv[c];
    float4 v = ((const float4*)(h + (size_t)r * F4 * 4))[threadIdx.x];
    float* op = out + (size_t)c * F4 * 4 + (size_t)threadIdx.x * 4;
    atomicAdd(op + 0, v.x * norm);
    atomicAdd(op + 1, v.y * norm);
    atomicAdd(op + 2, v.z * norm);
    atomicAdd(op + 3, v.w * norm);
}

// ---------------------------------------------------------------------------
extern "C" void kernel_launch(void* const* d_in, const int* in_sizes, int n_in,
                              void* d_out, int out_size, void* d_ws, size_t ws_size,
                              hipStream_t stream)
{
    const float* x        = (const float*)d_in[0];
    const int*   sim_ei   = (const int*)d_in[1];
    const float* sim_ew   = (const float*)d_in[2];
    const int*   dist_ei  = (const int*)d_in[3];
    const float* dist_ew  = (const float*)d_in[4];
    const float* W1       = (const float*)d_in[5];
    const float* b1       = (const float*)d_in[6];
    const float* W2       = (const float*)d_in[7];
    const float* b2       = (const float*)d_in[8];
    const float* Ws       = (const float*)d_in[9];
    const float* bs       = (const float*)d_in[10];
    const float* Wd       = (const float*)d_in[11];
    const float* bd       = (const float*)d_in[12];
    const float* Wf       = (const float*)d_in[13];
    const float* bfv      = (const float*)d_in[14];

    const int H   = in_sizes[6];                 // 512
    const int O   = in_sizes[10];                // 256
    const int Fin = in_sizes[5] / H;             // 3000
    const int N   = (int)(in_sizes[0] / Fin);    // 20000
    const int E   = in_sizes[2];                 // 640000
    const int Kp  = ((Fin + 31) / 32) * 32;      // 3008 (pad for BK=32)

    // ---- workspace layout ----
    size_t off = 0;
    auto alloc = [&](size_t bytes) -> char* {
        size_t o = off;
        off = (off + bytes + 255) & ~(size_t)255;
        return (char*)d_ws + o;
    };
    __bf16* xbf   = (__bf16*)alloc((size_t)N * Kp * 2);
    __bf16* w1bf  = (__bf16*)alloc((size_t)Kp * H * 2);
    __bf16* w2bf  = (__bf16*)alloc((size_t)Kp * H * 2);
    __bf16* wsbf  = (__bf16*)alloc((size_t)H * O * 2);
    __bf16* wdbf  = (__bf16*)alloc((size_t)H * O * 2);
    __bf16* wfbf  = (__bf16*)alloc((size_t)(2 * O) * O * 2);
    float*  h1    = (float*)alloc((size_t)N * H * 4);
    float*  h2    = (float*)alloc((size_t)N * H * 4);
    float*  a1s   = (float*)alloc((size_t)N * H * 4);
    float*  a1d   = (float*)alloc((size_t)N * H * 4);
    __bf16* xsbf  = (__bf16*)alloc((size_t)N * H * 2);
    __bf16* xdbf  = (__bf16*)alloc((size_t)N * H * 2);
    float*  hs2   = (float*)alloc((size_t)N * O * 4);
    float*  hd2   = (float*)alloc((size_t)N * O * 4);
    __bf16* catbf = (__bf16*)alloc((size_t)N * 2 * O * 2);
    float*  deg_s = (float*)alloc((size_t)N * 4);
    float*  deg_d = (float*)alloc((size_t)N * 4);
    float*  dinv_s= (float*)alloc((size_t)N * 4);
    float*  dinv_d= (float*)alloc((size_t)N * 4);
    (void)ws_size; (void)n_in; (void)out_size;

    float* x_sim  = (float*)d_out;                        // [N, O]
    float* x_dist = (float*)d_out + (size_t)N * O;        // [N, O]
    float* fused  = (float*)d_out + (size_t)2 * N * O;    // [N, O]

    auto cdiv = [](size_t a, size_t b) { return (unsigned)((a + b - 1) / b); };

    // ---- 1. precision conversion (+ K padding) ----
    cvt_pad_bf16<<<cdiv((size_t)N * Kp, 256), 256, 0, stream>>>(x, xbf, N, N, Fin, Kp, 0);
    cvt_pad_bf16<<<cdiv((size_t)Kp * H, 256), 256, 0, stream>>>(W1, w1bf, Kp, Fin, H, H, 0);
    cvt_pad_bf16<<<cdiv((size_t)Kp * H, 256), 256, 0, stream>>>(W2, w2bf, Kp, Fin, H, H, 0);
    cvt_pad_bf16<<<cdiv((size_t)H * O, 256), 256, 0, stream>>>(Ws, wsbf, H, H, O, O, 0);
    cvt_pad_bf16<<<cdiv((size_t)H * O, 256), 256, 0, stream>>>(Wd, wdbf, H, H, O, O, 0);
    cvt_pad_bf16<<<cdiv((size_t)2 * O * O, 256), 256, 0, stream>>>(Wf, wfbf, 2 * O, 2 * O, O, O, 0);

    // ---- 2. degrees / norms (self-loop weight 1 folded into init) ----
    fill_ones<<<cdiv(N, 256), 256, 0, stream>>>(deg_s, N);
    fill_ones<<<cdiv(N, 256), 256, 0, stream>>>(deg_d, N);
    deg_acc<<<cdiv(E, 256), 256, 0, stream>>>(sim_ei + E, sim_ew, deg_s, E);
    deg_acc<<<cdiv(E, 256), 256, 0, stream>>>(dist_ei + E, dist_ew, deg_d, E);
    make_dinv<<<cdiv(N, 256), 256, 0, stream>>>(deg_s, dinv_s, N);
    make_dinv<<<cdiv(N, 256), 256, 0, stream>>>(deg_d, dinv_d, N);

    // ---- 3. layer-1 GEMMs: h = x @ W ----
    {
        dim3 g(H / BN, cdiv(N, BM));
        gemm_bf16_wmma<<<g, 256, 0, stream>>>(xbf, w1bf, h1, nullptr, N, H, Kp);
        gemm_bf16_wmma<<<g, 256, 0, stream>>>(xbf, w2bf, h2, nullptr, N, H, Kp);
    }

    // ---- 4. layer-1 aggregation: out = b + h*dinv^2 + scatter(edges) ----
    const int H4 = H / 4;
    agg_init<<<cdiv((size_t)N * H4, 256), 256, 0, stream>>>(h1, dinv_s, b1, a1s, N, H4);
    agg_init<<<cdiv((size_t)N * H4, 256), 256, 0, stream>>>(h2, dinv_d, b2, a1d, N, H4);
    agg_scatter<<<E, H4, 0, stream>>>(h1, sim_ei, sim_ei + E, sim_ew, dinv_s, a1s, E, H4);
    agg_scatter<<<E, H4, 0, stream>>>(h2, dist_ei, dist_ei + E, dist_ew, dinv_d, a1d, E, H4);

    // ---- 5. ReLU + bf16 for layer 2 ----
    cvt_pad_bf16<<<cdiv((size_t)N * H, 256), 256, 0, stream>>>(a1s, xsbf, N, N, H, H, 1);
    cvt_pad_bf16<<<cdiv((size_t)N * H, 256), 256, 0, stream>>>(a1d, xdbf, N, N, H, H, 1);

    // ---- 6. layer-2 GEMMs ----
    {
        dim3 g(O / BN, cdiv(N, BM));
        gemm_bf16_wmma<<<g, 256, 0, stream>>>(xsbf, wsbf, hs2, nullptr, N, O, H);
        gemm_bf16_wmma<<<g, 256, 0, stream>>>(xdbf, wdbf, hd2, nullptr, N, O, H);
    }

    // ---- 7. layer-2 aggregation -> d_out (x_sim, x_dist) ----
    const int O4 = O / 4;
    agg_init<<<cdiv((size_t)N * O4, 256), 256, 0, stream>>>(hs2, dinv_s, bs, x_sim, N, O4);
    agg_init<<<cdiv((size_t)N * O4, 256), 256, 0, stream>>>(hd2, dinv_d, bd, x_dist, N, O4);
    agg_scatter<<<E, O4, 0, stream>>>(hs2, sim_ei, sim_ei + E, sim_ew, dinv_s, x_sim, E, O4);
    agg_scatter<<<E, O4, 0, stream>>>(hd2, dist_ei, dist_ei + E, dist_ew, dinv_d, x_dist, E, O4);

    // ---- 8. fused = concat(x_sim, x_dist) @ Wf + bf ----
    cvt_cat_bf16<<<cdiv((size_t)N * 2 * O, 256), 256, 0, stream>>>(x_sim, x_dist, catbf, N, O);
    {
        dim3 g(O / BN, cdiv(N, BM));
        gemm_bf16_wmma<<<g, 256, 0, stream>>>(catbf, wfbf, fused, bfv, N, O, 2 * O);
    }
}
